// Recurrence_38190849196657
// MI455X (gfx1250) — matrix operand order repaired and checked by reading, not compiled, not run on hardware
//
#include <hip/hip_runtime.h>

typedef __attribute__((ext_vector_type(16))) _Float16 v16h;
typedef __attribute__((ext_vector_type(8)))  float    v8f;

#define Hh   4
#define Kk   128
#define Ss   64
#define HID  512
#define NAa  16
#define Dd   64
#define Tt   128
#define Nn   32
#define TOT  25746

#define NT      512   // threads per block
#define NWAVES  16

// hx0 / y field offsets (floats)
#define OFF_AP   1
#define OFF_VAL  17
#define OFF_R    18
#define OFF_WR   274
#define OFF_U    786
#define OFF_WW   914
#define OFF_M    1042
#define OFF_P    9234
#define OFF_L    9362

#define MP 65    // M row pitch (pad to avoid bank conflicts)
#define LP 129   // L row pitch

// ssmall indices
#define SBR   0   // 4
#define SFR   4   // 4
#define SBW   8
#define SGA   9
#define SGW   10
#define SPI   11  // 12
#define SNKW  23
#define SNKR  24  // 4
#define SVAL  29

__device__ __forceinline__ float sigmoidf_(float x) { return 1.0f / (1.0f + __expf(-x)); }
__device__ __forceinline__ float softplusf_(float x) {
  float ax = fabsf(x);
  return fmaxf(x, 0.0f) + log1pf(__expf(-ax));
}

// ---------------------------------------------------------------------------
// Pack fp32 row-major weight [R x C] into f16 WMMA A-fragment tiles.
// A 16-bit 16x32 layout: lane m = lane&15; VGPR j (pair e=2j,2j+1):
//   kbase = (j<4 ? 2j : 16+2(j-4)) + ((lane&16)?8:0); k = kc*32 + kbase + (e&1)
// ---------------------------------------------------------------------------
__global__ void pack_weights(const float* __restrict__ W, _Float16* __restrict__ dst,
                             int R, int C, int Rp, int Cp) {
  int KcN = Cp >> 5;
  int nmc = Rp >> 4;
  size_t total = (size_t)nmc * KcN * 512;
  for (size_t i = (size_t)blockIdx.x * blockDim.x + threadIdx.x; i < total;
       i += (size_t)gridDim.x * blockDim.x) {
    int e    = (int)(i & 15);
    int lane = (int)((i >> 4) & 31);
    size_t tile = i >> 9;
    int kc = (int)(tile % KcN);
    int mc = (int)(tile / KcN);
    int vg = e >> 1, hf = e & 1;
    int kbase = (vg < 4 ? 2 * vg : 16 + 2 * (vg - 4)) + ((lane & 16) ? 8 : 0);
    int k = kc * 32 + kbase + hf;
    int m = mc * 16 + (lane & 15);
    float v = (m < R && k < C) ? W[(size_t)m * C + k] : 0.0f;
    dst[i] = (_Float16)v;
  }
}

__global__ void pad_bias(const float* __restrict__ b, float* __restrict__ dst, int R, int Rp) {
  int i = blockIdx.x * blockDim.x + threadIdx.x;
  if (i < Rp) dst[i] = (i < R) ? b[i] : 0.0f;
}

// ---------------------------------------------------------------------------
// GEMV via v_wmma_f32_16x16x32_f16, x broadcast across B columns.
// nmc == CPW*NWAVES always (weights padded): each wave owns exactly CPW output
// chunks -> CPW resident accumulator chains. kc loop is a REAL loop (unroll
// disabled) so liveness stays bounded (no scratch spills); within an iteration
// the CPW independent A-tile loads are clustered ahead of their WMMAs.
// No divergent branches around WMMAs (EXEC all-ones, per ISA).
// ---------------------------------------------------------------------------
template <int KC, int CPW>
__device__ __forceinline__ void gemv_wmma(const _Float16* __restrict__ Wp,
                                          const float* __restrict__ x,
                                          const float* __restrict__ bias,
                                          float* __restrict__ out,
                                          int relu, int lane, int wave) {
  v8f acc[CPW];
  v8f zero = {};
#pragma unroll
  for (int c = 0; c < CPW; ++c) acc[c] = zero;
  const _Float16* tb[CPW];
#pragma unroll
  for (int c = 0; c < CPW; ++c)
    tb[c] = Wp + ((size_t)(wave + c * NWAVES) * KC) * 512 + lane * 16;

  const float* xb = x + ((lane & 16) ? 8 : 0);

#pragma clang loop unroll(disable)
  for (int kc = 0; kc < KC; ++kc) {
    // B fragment for this k-slab from LDS: 4 independent b128 loads, then cvt
    const float4* x4 = (const float4*)(xb + kc * 32);
    float4 q0 = x4[0], q1 = x4[1], q2 = x4[4], q3 = x4[5];
    v16h bf;
    bf[0]  = (_Float16)q0.x; bf[1]  = (_Float16)q0.y;
    bf[2]  = (_Float16)q0.z; bf[3]  = (_Float16)q0.w;
    bf[4]  = (_Float16)q1.x; bf[5]  = (_Float16)q1.y;
    bf[6]  = (_Float16)q1.z; bf[7]  = (_Float16)q1.w;
    bf[8]  = (_Float16)q2.x; bf[9]  = (_Float16)q2.y;
    bf[10] = (_Float16)q2.z; bf[11] = (_Float16)q2.w;
    bf[12] = (_Float16)q3.x; bf[13] = (_Float16)q3.y;
    bf[14] = (_Float16)q3.z; bf[15] = (_Float16)q3.w;
    // clustered A-tile loads, then CPW independent WMMA chains
    v16h a[CPW];
#pragma unroll
    for (int c = 0; c < CPW; ++c)
      a[c] = *(const v16h*)(tb[c] + (size_t)kc * 512);
#pragma unroll
    for (int c = 0; c < CPW; ++c)
      acc[c] = __builtin_amdgcn_wmma_f32_16x16x32_f16(false, a[c], false, bf,
                                                      (short)0, acc[c], false, false);
  }

  if ((lane & 15) == 0) {
    int half = (lane & 16) ? 8 : 0;
#pragma unroll
    for (int c = 0; c < CPW; ++c) {
      int mb = (wave + c * NWAVES) * 16 + half;
#pragma unroll
      for (int q = 0; q < 8; ++q) {
        float o = acc[c][q] + bias[mb + q];
        if (relu) o = fmaxf(o, 0.0f);
        out[mb + q] = o;
      }
    }
  }
}

// ---------------------------------------------------------------------------
// Main recurrence: one workgroup (512 threads = 16 waves) per batch element.
// All carried state (M, L, r, wr, u, ww, p) LDS-resident for all 128 steps.
// ---------------------------------------------------------------------------
__global__ __launch_bounds__(NT, 1)
void dnc_recurrence(const float* __restrict__ obs, const int* __restrict__ actions,
                    const float* __restrict__ hx0,
                    const float* __restrict__ bf1, const float* __restrict__ bih,
                    const float* __restrict__ bhh, const float* __restrict__ bf2p,
                    const float* __restrict__ Wa,  const float* __restrict__ ba,
                    const float* __restrict__ Wc,  const float* __restrict__ bc,
                    const _Float16* __restrict__ pWf1, const _Float16* __restrict__ pWih,
                    const _Float16* __restrict__ pWhh, const _Float16* __restrict__ pWf2,
                    float* __restrict__ y_out) {
  __shared__ float sL[Kk * LP];        // 66 KB
  __shared__ float sM[Kk * MP];        // 33 KB
  __shared__ float sr[256], srf[256];
  __shared__ float sh0[HID], shrelu[HID];
  __shared__ float sgi[3 * HID], sgh[3 * HID];
  __shared__ float sx[Dd];
  __shared__ float sxi[512];
  __shared__ float swr[Hh * Kk], swrn[Hh * Kk];
  __shared__ float su[Kk], sun[Kk], sww[Kk], swwn[Kk], sp[Kk];
  __shared__ float salloc[Kk], scw[Kk], sup[Kk], scp[Kk];
  __shared__ float skey[Kk];
  __shared__ int   sidx[Kk];
  __shared__ float sb[Hh * Kk], sf[Hh * Kk], scr[Hh * Kk];
  __shared__ float sdot[Kk];
  __shared__ float se[Ss], sv[Ss], skw[Ss];
  __shared__ float sred[NT];
  __shared__ float sprob[NAa];
  __shared__ float ssmall[32];

  const int n = blockIdx.x;
  const int tid = threadIdx.x;
  const int lane = tid & 31;
  const int wave = __builtin_amdgcn_readfirstlane(tid >> 5);  // SGPR wave id

  // ---- load initial carry from hx0[0][n][:] ----
  const float* hxb = hx0 + (size_t)n * TOT;
  for (int i = tid; i < 256; i += NT) sr[i] = hxb[OFF_R + i];
  for (int i = tid; i < 512; i += NT) swr[i] = hxb[OFF_WR + i];
  if (tid < Kk) {
    su[tid]  = hxb[OFF_U + tid];
    sww[tid] = hxb[OFF_WW + tid];
    sp[tid]  = hxb[OFF_P + tid];
  }
  for (int i = tid; i < Kk * Ss; i += NT) {
    int k = i >> 6, s = i & 63;
    sM[k * MP + s] = hxb[OFF_M + i];
  }
  for (int i = tid; i < Kk * Kk; i += NT) {
    int k = i >> 7, j = i & 127;
    sL[k * LP + j] = hxb[OFF_L + i];
  }

  for (int t = 0; t < Tt; ++t) {
    __syncthreads();
    if (tid < Dd) sx[tid] = obs[((size_t)t * Nn + n) * Dd + tid];
    __syncthreads();

    // ---- controller: h0 = relu(Wf1 r + bf1) ----
    gemv_wmma<8, 2>(pWf1, sr, bf1, sh0, 1, lane, wave);
    __syncthreads();
    // ---- GRU gates ----
    gemv_wmma<2, 6>(pWih, sx, bih, sgi, 0, lane, wave);
    gemv_wmma<16, 6>(pWhh, sh0, bhh, sgh, 0, lane, wave);
    __syncthreads();
    for (int i = tid; i < HID; i += NT) {
      float rg = sigmoidf_(sgi[i] + sgh[i]);
      float zg = sigmoidf_(sgi[HID + i] + sgh[HID + i]);
      float ng = tanhf(sgi[2 * HID + i] + rg * sgh[2 * HID + i]);
      float hh = (1.0f - zg) * ng + zg * sh0[i];
      shrelu[i] = fmaxf(hh, 0.0f);
    }
    __syncthreads();
    // ---- interface vector xi = Wf2 relu(h) + bf2 (padded 471 -> 512 rows) ----
    gemv_wmma<16, 2>(pWf2, shrelu, bf2p, sxi, 0, lane, wave);
    __syncthreads();

    // ---- xi activations ----
    if (tid < Ss) {
      skw[tid] = sxi[260 + tid];
      se[tid]  = sigmoidf_(sxi[325 + tid]);
      sv[tid]  = sxi[389 + tid];
    } else if (tid >= 64 && tid < 68) {
      int h = tid - 64;
      ssmall[SBR + h] = softplusf_(sxi[256 + h]);
      ssmall[SFR + h] = sigmoidf_(sxi[453 + h]);
    } else if (tid == 68) {
      ssmall[SBW] = softplusf_(sxi[324]);
      ssmall[SGA] = sigmoidf_(sxi[457]);
      ssmall[SGW] = sigmoidf_(sxi[458]);
    } else if (tid >= 72 && tid < 76) {
      int h = tid - 72;
      float p0 = sxi[459 + 3 * h], p1 = sxi[460 + 3 * h], p2 = sxi[461 + 3 * h];
      float mx = fmaxf(p0, fmaxf(p1, p2));
      float e0 = __expf(p0 - mx), e1 = __expf(p1 - mx), e2 = __expf(p2 - mx);
      float s = e0 + e1 + e2;
      ssmall[SPI + 3 * h + 0] = e0 / s;
      ssmall[SPI + 3 * h + 1] = e1 / s;
      ssmall[SPI + 3 * h + 2] = e2 / s;
    }
    __syncthreads();
    if (tid < Hh) {
      float s = 0.0f;
      for (int q = 0; q < Ss; ++q) { float a = sxi[tid * Ss + q]; s += a * a; }
      ssmall[SNKR + tid] = sqrtf(s);
    } else if (tid == 4) {
      float s = 0.0f;
      for (int q = 0; q < Ss; ++q) s += skw[q] * skw[q];
      ssmall[SNKW] = sqrtf(s);
    }
    __syncthreads();

    // ---- usage update + allocation (sort) ----
    if (tid < Kk) {
      float ps = 1.0f;
#pragma unroll
      for (int h = 0; h < Hh; ++h) ps *= (1.0f - ssmall[SFR + h] * swr[h * Kk + tid]);
      float uo = su[tid];
      float un = (uo + (1.0f - uo) * sww[tid]) * ps;
      sun[tid] = un;
      skey[tid] = un;
      sidx[tid] = tid;
    }
    // bitonic sort ascending on skey (payload sidx)
    for (int ksz = 2; ksz <= Kk; ksz <<= 1) {
      for (int jsz = ksz >> 1; jsz > 0; jsz >>= 1) {
        __syncthreads();
        if (tid < Kk) {
          int ixj = tid ^ jsz;
          if (ixj > tid) {
            bool asc = ((tid & ksz) == 0);
            float a = skey[tid], b = skey[ixj];
            bool sw = asc ? (a > b) : (a < b);
            if (sw) {
              skey[tid] = b; skey[ixj] = a;
              int ti = sidx[tid]; sidx[tid] = sidx[ixj]; sidx[ixj] = ti;
            }
          }
        }
      }
    }
    __syncthreads();
    if (tid == 0) {  // cumprod over sorted usage
      float c = 1.0f;
      for (int i = 0; i < Kk; ++i) { c *= skey[i]; scp[i] = c; }
    }
    __syncthreads();
    if (tid < Kk) sup[sidx[tid]] = scp[tid];
    __syncthreads();
    if (tid < Kk) salloc[tid] = (1.0f - sun[tid]) * sup[tid];

    // ---- write content weighting cw = softmax(bw * cos(M, kw)) (old M) ----
    if (tid < Kk) {
      float d = 0.0f, nm = 0.0f;
      for (int s = 0; s < Ss; ++s) {
        float mv = sM[tid * MP + s];
        d += mv * skw[s];
        nm += mv * mv;
      }
      nm = sqrtf(nm);
      float cosv = d / (fmaxf(nm, 1e-8f) * fmaxf(ssmall[SNKW], 1e-8f));
      sdot[tid] = ssmall[SBW] * cosv;
    }
    __syncthreads();
    sred[tid] = (tid < Kk) ? sdot[tid] : -3.4e38f;
    __syncthreads();
    for (int s2 = NT / 2; s2 >= 1; s2 >>= 1) {
      if (tid < s2) sred[tid] = fmaxf(sred[tid], sred[tid + s2]);
      __syncthreads();
    }
    float mx = sred[0];
    __syncthreads();
    if (tid < Kk) scw[tid] = __expf(sdot[tid] - mx);
    __syncthreads();
    sred[tid] = (tid < Kk) ? scw[tid] : 0.0f;
    __syncthreads();
    for (int s2 = NT / 2; s2 >= 1; s2 >>= 1) {
      if (tid < s2) sred[tid] += sred[tid + s2];
      __syncthreads();
    }
    float smsum = sred[0];
    __syncthreads();
    if (tid < Kk) {
      scw[tid] /= smsum;
      swwn[tid] = ssmall[SGW] * (ssmall[SGA] * salloc[tid] + (1.0f - ssmall[SGA]) * scw[tid]);
    }
    __syncthreads();

    // ---- memory write: M update ----
    for (int i = tid; i < Kk * Ss; i += NT) {
      int k = i >> 6, s = i & 63;
      float w = swwn[k];
      sM[k * MP + s] = sM[k * MP + s] * (1.0f - w * se[s]) + w * sv[s];
    }
    // sum(ww)
    sred[tid] = (tid < Kk) ? swwn[tid] : 0.0f;
    __syncthreads();
    for (int s2 = NT / 2; s2 >= 1; s2 >>= 1) {
      if (tid < s2) sred[tid] += sred[tid + s2];
      __syncthreads();
    }
    float sumww = sred[0];
    __syncthreads();
    if (tid < Kk) sp[tid] = (1.0f - sumww) * sp[tid] + swwn[tid];
    __syncthreads();
    // ---- temporal linkage L update ----
    for (int i = tid; i < Kk * Kk; i += NT) {
      int k = i >> 7, j = i & 127;
      float lv = (k == j) ? 0.0f
                          : (1.0f - swwn[k] - swwn[j]) * sL[k * LP + j] + swwn[k] * sp[k];
      sL[k * LP + j] = lv;
    }
    __syncthreads();

    // ---- b = wr L (threads 0-127, column pass), f = wr L^T (threads 128-255, row pass)
    if (tid < Kk) {
      float b0 = 0, b1 = 0, b2 = 0, b3 = 0;
      for (int k = 0; k < Kk; ++k) {
        float lkj = sL[k * LP + tid];
        b0 += swr[0 * Kk + k] * lkj;
        b1 += swr[1 * Kk + k] * lkj;
        b2 += swr[2 * Kk + k] * lkj;
        b3 += swr[3 * Kk + k] * lkj;
      }
      sb[0 * Kk + tid] = b0; sb[1 * Kk + tid] = b1;
      sb[2 * Kk + tid] = b2; sb[3 * Kk + tid] = b3;
    } else if (tid < 2 * Kk) {
      int j = tid - Kk;
      float f0 = 0, f1 = 0, f2 = 0, f3 = 0;
      for (int k = 0; k < Kk; ++k) {
        float ljk = sL[j * LP + k];
        f0 += swr[0 * Kk + k] * ljk;
        f1 += swr[1 * Kk + k] * ljk;
        f2 += swr[2 * Kk + k] * ljk;
        f3 += swr[3 * Kk + k] * ljk;
      }
      sf[0 * Kk + j] = f0; sf[1 * Kk + j] = f1;
      sf[2 * Kk + j] = f2; sf[3 * Kk + j] = f3;
    }

    // ---- read content weighting cr (new M) ----
    if (tid < Kk) {
      float nm = 0, d0 = 0, d1 = 0, d2 = 0, d3 = 0;
      for (int s = 0; s < Ss; ++s) {
        float mv = sM[tid * MP + s];
        nm += mv * mv;
        d0 += mv * sxi[0 * Ss + s];
        d1 += mv * sxi[1 * Ss + s];
        d2 += mv * sxi[2 * Ss + s];
        d3 += mv * sxi[3 * Ss + s];
      }
      nm = fmaxf(sqrtf(nm), 1e-8f);
      scr[0 * Kk + tid] = ssmall[SBR + 0] * d0 / (nm * fmaxf(ssmall[SNKR + 0], 1e-8f));
      scr[1 * Kk + tid] = ssmall[SBR + 1] * d1 / (nm * fmaxf(ssmall[SNKR + 1], 1e-8f));
      scr[2 * Kk + tid] = ssmall[SBR + 2] * d2 / (nm * fmaxf(ssmall[SNKR + 2], 1e-8f));
      scr[3 * Kk + tid] = ssmall[SBR + 3] * d3 / (nm * fmaxf(ssmall[SNKR + 3], 1e-8f));
    }
    __syncthreads();
    if (tid < Hh) {  // per-head softmax over K
      float m2 = -3.4e38f;
      for (int k = 0; k < Kk; ++k) m2 = fmaxf(m2, scr[tid * Kk + k]);
      float s = 0.0f;
      for (int k = 0; k < Kk; ++k) {
        float e2 = __expf(scr[tid * Kk + k] - m2);
        scr[tid * Kk + k] = e2;
        s += e2;
      }
      float inv = 1.0f / s;
      for (int k = 0; k < Kk; ++k) scr[tid * Kk + k] *= inv;
    }
    __syncthreads();
    // ---- new read weights ----
    if (tid < Kk) {
#pragma unroll
      for (int h = 0; h < Hh; ++h) {
        swrn[h * Kk + tid] = ssmall[SPI + 3 * h + 0] * sb[h * Kk + tid] +
                             ssmall[SPI + 3 * h + 1] * scr[h * Kk + tid] +
                             ssmall[SPI + 3 * h + 2] * sf[h * Kk + tid];
      }
    }
    __syncthreads();
    // ---- read vectors rf = wr M ----
    for (int i = tid; i < Hh * Ss; i += NT) {
      int h = i >> 6, s = i & 63;
      float acc = 0.0f;
      for (int k = 0; k < Kk; ++k) acc += swrn[h * Kk + k] * sM[k * MP + s];
      srf[i] = acc;
    }
    __syncthreads();
    // ---- heads ----
    if (tid < NAa) {
      float acc = ba[tid];
      for (int i = 0; i < 256; ++i) acc += Wa[tid * 256 + i] * srf[i];
      sprob[tid] = acc;
    } else if (tid == 16) {
      float acc = bc[0];
      for (int i = 0; i < 256; ++i) acc += Wc[i] * srf[i];
      ssmall[SVAL] = acc;
    }
    __syncthreads();
    if (tid == 0) {
      float m2 = -3.4e38f;
      for (int a = 0; a < NAa; ++a) m2 = fmaxf(m2, sprob[a]);
      float s = 0.0f;
      for (int a = 0; a < NAa; ++a) { float e2 = __expf(sprob[a] - m2); sprob[a] = e2; s += e2; }
      float inv = 1.0f / s;
      for (int a = 0; a < NAa; ++a) sprob[a] *= inv;
    }
    __syncthreads();

    // ---- emit y[t][n][:] (non-temporal: write-once stream, keep L2 for weights)
    float* y = y_out + ((size_t)t * Nn + n) * TOT;
    if (tid == 0) {
      __builtin_nontemporal_store((float)actions[(size_t)t * Nn + n], &y[0]);
      __builtin_nontemporal_store(ssmall[SVAL], &y[OFF_VAL]);
    }
    if (tid < NAa) __builtin_nontemporal_store(sprob[tid], &y[OFF_AP + tid]);
    for (int i = tid; i < 256; i += NT) __builtin_nontemporal_store(srf[i], &y[OFF_R + i]);
    for (int i = tid; i < 512; i += NT) __builtin_nontemporal_store(swrn[i], &y[OFF_WR + i]);
    if (tid < Kk) {
      __builtin_nontemporal_store(sun[tid],  &y[OFF_U + tid]);
      __builtin_nontemporal_store(swwn[tid], &y[OFF_WW + tid]);
      __builtin_nontemporal_store(sp[tid],   &y[OFF_P + tid]);
    }
    for (int i = tid; i < Kk * Ss; i += NT) {
      int k = i >> 6, s = i & 63;
      __builtin_nontemporal_store(sM[k * MP + s], &y[OFF_M + i]);
    }
    for (int i = tid; i < Kk * Kk; i += NT) {
      int k = i >> 7, j = i & 127;
      __builtin_nontemporal_store(sL[k * LP + j], &y[OFF_L + i]);
    }
    __syncthreads();
    // ---- carry over ----
    for (int i = tid; i < 256; i += NT) sr[i] = srf[i];
    for (int i = tid; i < 512; i += NT) swr[i] = swrn[i];
    if (tid < Kk) { su[tid] = sun[tid]; sww[tid] = swwn[tid]; }
  }
}

// ---------------------------------------------------------------------------
extern "C" void kernel_launch(void* const* d_in, const int* in_sizes, int n_in,
                              void* d_out, int out_size, void* d_ws, size_t ws_size,
                              hipStream_t stream) {
  const float* obs     = (const float*)d_in[0];
  const int*   actions = (const int*)d_in[1];
  const float* hx0     = (const float*)d_in[2];
  const float* Wf1 = (const float*)d_in[3];
  const float* bf1 = (const float*)d_in[4];
  const float* Wih = (const float*)d_in[5];
  const float* Whh = (const float*)d_in[6];
  const float* bih = (const float*)d_in[7];
  const float* bhh = (const float*)d_in[8];
  const float* Wf2 = (const float*)d_in[9];
  const float* bf2 = (const float*)d_in[10];
  const float* Wa  = (const float*)d_in[11];
  const float* ba  = (const float*)d_in[12];
  const float* Wc  = (const float*)d_in[13];
  const float* bc  = (const float*)d_in[14];

  _Float16* ws = (_Float16*)d_ws;
  _Float16* pWf1 = ws;              // 32*8   tiles = 131072 halves
  _Float16* pWih = ws + 131072;     // 96*2   tiles =  98304 halves
  _Float16* pWhh = ws + 229376;     // 96*16  tiles = 786432 halves
  _Float16* pWf2 = ws + 1015808;    // 32*16  tiles = 262144 halves (rows padded to 512)
  float*    bf2p = (float*)(ws + 1277952);  // 512 padded floats

  pack_weights<<<256,  256, 0, stream>>>(Wf1, pWf1, 512, 256, 512, 256);
  pack_weights<<<192,  256, 0, stream>>>(Wih, pWih, 1536, 64, 1536, 64);
  pack_weights<<<1024, 256, 0, stream>>>(Whh, pWhh, 1536, 512, 1536, 512);
  pack_weights<<<512,  256, 0, stream>>>(Wf2, pWf2, 471, 512, 512, 512);
  pad_bias<<<2, 256, 0, stream>>>(bf2, bf2p, 471, 512);

  dnc_recurrence<<<Nn, NT, 0, stream>>>(obs, actions, hx0,
                                        bf1, bih, bhh, bf2p, Wa, ba, Wc, bc,
                                        pWf1, pWih, pWhh, pWf2, (float*)d_out);
}
